// AttentionModel_50139448214008
// MI455X (gfx1250) — matrix operand order repaired and checked by reading, not compiled, not run on hardware
//
#include <hip/hip_runtime.h>
#include <hip/hip_bf16.h>
#include <math.h>

// Problem constants (match reference)
#define B_  128
#define N_  2000
#define D_  128
#define H_  8
#define HD_ 16
#define MT  64   // rows of embeddings per proj block

typedef __attribute__((ext_vector_type(16))) _Float16 v16h;
typedef __attribute__((ext_vector_type(8)))  _Float16 v8h;
typedef __attribute__((ext_vector_type(4)))  _Float16 v4h;
typedef __attribute__((ext_vector_type(8)))  float    v8f;
typedef __attribute__((ext_vector_type(4)))  unsigned int u32x4;
typedef __attribute__((ext_vector_type(8)))  int      i32x8;
typedef __attribute__((ext_vector_type(4)))  int      i32x4;

// ---------------------------------------------------------------------------
// TDM helper: issue TENSOR_LOAD_TO_LDS for a 2-D f16 tile (row-major).
//  gsrc       : global source (f16), start of tile
//  lds_off    : LDS byte offset of destination
//  tile_w     : tile width in elements (dim0)
//  tile_h     : tile height in rows (dim1)
//  row_stride : tensor row stride in elements
// Descriptor layout per CDNA5 ISA:
//  group0: [1:0]=count=1, [63:32]=lds_addr, [120:64]=global_addr, [127:126]=2
//  group1: [17:16]=data_size(1 => 2B), [79:48]=tensor_dim0, [111:80]=tensor_dim1,
//          [127:112]=tile_dim0, [143:128]=tile_dim1, [207:160]=tensor_dim0_stride
//  trailing groups: zero (tile_dim2..4 = 0 -> 2-D tile)
// Tracked by TENSORcnt; caller waits with s_wait_tensorcnt.
// This toolchain exposes the 6-arg builtin:
//   (u32x4 g0, i32x8 g1, i32x4, i32x4, i32x8, i32 cpol)
// ---------------------------------------------------------------------------
__device__ inline void tdm_load_2d_f16(const _Float16* gsrc, unsigned lds_off,
                                       int tile_w, int tile_h, int row_stride) {
  unsigned long long ga = (unsigned long long)(size_t)gsrc;
  u32x4 g0;
  g0[0] = 1u;                                              // count=1, user mode
  g0[1] = lds_off;                                         // lds_addr
  g0[2] = (unsigned)(ga & 0xFFFFFFFFu);                    // global_addr[31:0]
  g0[3] = (unsigned)((ga >> 32) & 0x1FFFFFFu) | (2u << 30);// ga[56:32] | type=2
  i32x8 g1;
  g1[0] = (1 << 16);                                       // data_size=1 (2B)
  g1[1] = (row_stride & 0xFFFF) << 16;                     // tensor_dim0[15:0]
  g1[2] = (tile_h & 0xFFFF) << 16;                         // tensor_dim1[15:0]
  g1[3] = (tile_w & 0xFFFF) << 16;                         // tile_dim0
  g1[4] = (tile_h & 0xFFFF);                               // tile_dim1 (dim2=0)
  g1[5] = row_stride;                                      // tensor_dim0_stride lo
  g1[6] = 0;
  g1[7] = 0;
  i32x4 gz4 = {0, 0, 0, 0};
  i32x8 gz8 = {0, 0, 0, 0, 0, 0, 0, 0};
  __builtin_amdgcn_tensor_load_to_lds(g0, g1, gz4, gz4, gz8, 0);
}

__device__ inline unsigned lds_offset_of(const void* p) {
  // generic LDS pointer = {shared aperture, 32-bit LDS offset}; truncate.
  return (unsigned)(size_t)p;
}

// ---------------------------------------------------------------------------
// Kernel 0: one streaming pass converting embeddings (fp32 -> f16, 131->64 MB)
// and the three projection weights to f16 in workspace. Removes all per-block
// conversion VALU from the GEMM kernel and halves its read traffic.
// ---------------------------------------------------------------------------
__global__ __launch_bounds__(256)
void cvt_kernel(const float* __restrict__ emb, _Float16* __restrict__ ef16,
                const float* __restrict__ Wk, const float* __restrict__ Wv,
                const float* __restrict__ Wkt, _Float16* __restrict__ wf16) {
  const int tid = threadIdx.x;
  const size_t total4 = (size_t)B_ * N_ * D_ / 4;
  const float4* src = (const float4*)emb;
  for (size_t i = (size_t)blockIdx.x * 256 + tid; i < total4;
       i += (size_t)gridDim.x * 256) {
    float4 f = src[i];
    v4h o = {(_Float16)f.x, (_Float16)f.y, (_Float16)f.z, (_Float16)f.w};
    *(v4h*)(ef16 + 4 * i) = o;
  }
  if (blockIdx.x < 3) {
    const float* W = (blockIdx.x == 0) ? Wk : ((blockIdx.x == 1) ? Wv : Wkt);
    const float4* ws4 = (const float4*)W;
    _Float16* dst = wf16 + (size_t)blockIdx.x * D_ * D_;
    for (int i = tid; i < D_ * D_ / 4; i += 256) {
      float4 f = ws4[i];
      v4h o = {(_Float16)f.x, (_Float16)f.y, (_Float16)f.z, (_Float16)f.w};
      *(v4h*)(dst + 4 * i) = o;
    }
  }
}

// ---------------------------------------------------------------------------
// Kernel 1: Q[b,d] = bq_c[d] + bq_s[d] + Wq_c[d,:]·ctx[b,:] + Wq_s[d,:]·step[b,:]
// ---------------------------------------------------------------------------
__global__ __launch_bounds__(128)
void q_kernel(const float* __restrict__ ctx, const float* __restrict__ stepc,
              const float* __restrict__ Wqc, const float* __restrict__ bqc,
              const float* __restrict__ Wqs, const float* __restrict__ bqs,
              float* __restrict__ Qout) {
  const int b = blockIdx.x, d = threadIdx.x;
  const float* c   = ctx   + (size_t)b * D_;
  const float* s   = stepc + (size_t)b * (D_ + 1);
  const float* wc  = Wqc   + (size_t)d * D_;
  const float* wsr = Wqs   + (size_t)d * (D_ + 1);
  float acc = bqc[d] + bqs[d];
  #pragma unroll 8
  for (int j = 0; j < D_; ++j)     acc += wc[j]  * c[j];
  #pragma unroll 8
  for (int j = 0; j < D_ + 1; ++j) acc += wsr[j] * s[j];
  Qout[(size_t)b * D_ + d] = acc;
}

// ---------------------------------------------------------------------------
// Kernel 2: fused K/V/K_tanh projection GEMM via v_wmma_f32_16x16x32_f16.
// Tiles arrive in LDS via TDM (TENSOR_LOAD_TO_LDS): one DMA for the
// 64x128 f16 embedding tile, one per 128x128 f16 weight, synchronized with
// s_wait_tensorcnt + barrier. 8 waves each own a 16-wide output column strip.
//
// Fragment layouts used here (contiguous k order, same for A and B):
//  A (16x32): lane<16 -> row=lane, K = kb..kb+15
//             lane>=16 -> row=lane-16, K = kb+16..kb+31
//  B (32x16): lane<16 -> col=lane,   K = kb..kb+15 (contiguous in W row)
//             lane>=16 -> col=lane-16, K = kb+16..kb+31
//  C (16x16 f32): vgpr j -> M = (lane>=16 ? 8 : 0)+j, N = lane&15
// ---------------------------------------------------------------------------
__global__ __launch_bounds__(256)
void proj_kernel(const _Float16* __restrict__ ef16,
                 const _Float16* __restrict__ wf16,
                 const float* __restrict__ bk, const float* __restrict__ bv,
                 const float* __restrict__ bkt,
                 float* __restrict__ Kout, float* __restrict__ Vout,
                 float* __restrict__ KTout) {
  __shared__ _Float16 sE[MT * D_];   // 16 KB, TDM destination
  __shared__ _Float16 sW[D_ * D_];   // 32 KB, TDM destination

  const int tid = threadIdx.x;
  const int m0  = blockIdx.x * MT;

  // Embedding tile DMA: issued once per block by wave 0.
  if (tid < 32) {
    tdm_load_2d_f16(ef16 + (size_t)m0 * D_, lds_offset_of(&sE[0]),
                    D_, MT, D_);
  }

  const int wave = tid >> 5;
  const int lane = tid & 31;
  const int half = lane >> 4;
  const int r    = lane & 15;
  const int nb   = wave * 16;   // output-column tile owned by this wave

  for (int w = 0; w < 3; ++w) {
    const float* bias = (w == 0) ? bk   : ((w == 1) ? bv   : bkt);
    float*       Out  = (w == 0) ? Kout : ((w == 1) ? Vout : KTout);

    __syncthreads();   // previous compute done reading sW
    if (tid < 32) {
      tdm_load_2d_f16(wf16 + (size_t)w * D_ * D_, lds_offset_of(&sW[0]),
                      D_, D_, D_);
      __builtin_amdgcn_s_wait_tensorcnt(0);   // emb tile + this weight done
    }
    __syncthreads();

    const float bval = bias[nb + r];
    for (int mt = 0; mt < 4; ++mt) {
      v8f acc = {};
      #pragma unroll
      for (int ks = 0; ks < 4; ++ks) {
        const int kb = ks * 32;
        // A fragment: two contiguous 8-half chunks per lane
        const _Float16* pa = &sE[(mt * 16 + r) * D_ + kb + half * 16];
        v8h a0 = *(const v8h*)(pa);
        v8h a1 = *(const v8h*)(pa + 8);
        v16h a = __builtin_shufflevector(a0, a1,
                   0,1,2,3,4,5,6,7,8,9,10,11,12,13,14,15);
        // B fragment: 16 contiguous halfs of W row (output col = W row)
        const _Float16* pb = &sW[(nb + r) * D_ + kb + half * 16];
        v16h bm = *(const v16h*)pb;
        acc = __builtin_amdgcn_wmma_f32_16x16x32_f16(
                  false, a, false, bm, (short)0, acc, false, false);
      }
      float* orow = Out + ((size_t)(m0 + mt * 16 + half * 8)) * D_ + nb + r;
      #pragma unroll
      for (int j = 0; j < 8; ++j) orow[(size_t)j * D_] = acc[j] + bval;
    }
  }
}

// ---------------------------------------------------------------------------
// Kernel 3: per-batch attention + output head. One 256-thread block per batch.
// Wave h performs online-softmax attention for head h over N=2000 keys
// (lane-strided, shuffle-merged, wave32). Then Wo GEMV, then logits GEMV
// against K_tanh rows with block-wide log-softmax.
// ---------------------------------------------------------------------------
__global__ __launch_bounds__(256)
void attn_kernel(const float* __restrict__ Kb, const float* __restrict__ Vb,
                 const float* __restrict__ KTb, const float* __restrict__ Qb,
                 const unsigned char* __restrict__ mask,  // jnp bool = 1 byte
                 const float* __restrict__ Wo, const float* __restrict__ bo,
                 float* __restrict__ out) {
  __shared__ float sQ[D_];
  __shared__ float sAttn[D_];
  __shared__ float sMha[D_];
  __shared__ float sLogit[N_];
  __shared__ float sRed[8];

  const int b    = blockIdx.x;
  const int tid  = threadIdx.x;
  const int wave = tid >> 5;
  const int lane = tid & 31;

  if (tid < D_) sQ[tid] = Qb[(size_t)b * D_ + tid];
  __syncthreads();

  // ---- online-softmax attention, head h = wave ----
  const int h = wave;
  float q[HD_];
  #pragma unroll
  for (int d = 0; d < HD_; ++d) q[d] = sQ[h * HD_ + d];

  const unsigned char* mrow = mask + (size_t)b * N_;
  const float* Kh = Kb + ((size_t)b * N_) * D_ + h * HD_;
  const float* Vh = Vb + ((size_t)b * N_) * D_ + h * HD_;

  float m = -__builtin_inff();
  float s = 0.f;
  float a[HD_];
  #pragma unroll
  for (int d = 0; d < HD_; ++d) a[d] = 0.f;

  for (int n = lane; n < N_; n += 32) {
    if (mrow[n]) continue;                 // masked -> contributes nothing
    const float* kr = Kh + (size_t)n * D_;
    float x = 0.f;
    #pragma unroll
    for (int d = 0; d < HD_; ++d) x += q[d] * kr[d];
    x *= 0.25f;                            // 1/sqrt(HD)
    float mn = fmaxf(m, x);
    float sc = __expf(m - mn);             // exp(-inf)=0 on first hit
    float fw = __expf(x - mn);
    const float* vr = Vh + (size_t)n * D_;
    s = s * sc + fw;
    #pragma unroll
    for (int d = 0; d < HD_; ++d) a[d] = a[d] * sc + fw * vr[d];
    m = mn;
  }
  // merge the 32 lane-local online-softmax states (wave32)
  float M = m;
  #pragma unroll
  for (int o = 16; o > 0; o >>= 1) M = fmaxf(M, __shfl_xor(M, o, 32));
  float f  = __expf(m - M);
  float st = s * f;
  #pragma unroll
  for (int o = 16; o > 0; o >>= 1) st += __shfl_xor(st, o, 32);
  float inv = 1.0f / st;
  #pragma unroll
  for (int d = 0; d < HD_; ++d) {
    float ad = a[d] * f;
    #pragma unroll
    for (int o = 16; o > 0; o >>= 1) ad += __shfl_xor(ad, o, 32);
    if (lane == 0) sAttn[h * HD_ + d] = ad * inv;
  }
  __syncthreads();

  // ---- mha = Wo @ attn + bo ----
  if (tid < D_) {
    const float* wr = Wo + (size_t)tid * D_;
    float acc = bo[tid];
    #pragma unroll 8
    for (int k = 0; k < D_; ++k) acc += wr[k] * sAttn[k];
    sMha[tid] = acc;
  }
  __syncthreads();

  // ---- logits[n] = mha . K_tanh[n] / sqrt(D), masked, then log_softmax ----
  const float* KT0 = KTb + ((size_t)b * N_) * D_;
  float lmax = -__builtin_inff();
  for (int n = tid; n < N_; n += 256) {
    float l;
    if (mrow[n]) {
      l = -__builtin_inff();
    } else {
      const float* kt = KT0 + (size_t)n * D_;
      float dot = 0.f;
      #pragma unroll 8
      for (int d = 0; d < D_; ++d) dot += sMha[d] * kt[d];
      l = dot * 0.08838834764831845f;      // 1/sqrt(128)
    }
    sLogit[n] = l;
    lmax = fmaxf(lmax, l);
  }
  #pragma unroll
  for (int o = 16; o > 0; o >>= 1) lmax = fmaxf(lmax, __shfl_xor(lmax, o, 32));
  if (lane == 0) sRed[wave] = lmax;
  __syncthreads();
  float gmax = sRed[0];
  for (int w2 = 1; w2 < 8; ++w2) gmax = fmaxf(gmax, sRed[w2]);
  __syncthreads();

  float lsum = 0.f;
  for (int n = tid; n < N_; n += 256) lsum += __expf(sLogit[n] - gmax);
  #pragma unroll
  for (int o = 16; o > 0; o >>= 1) lsum += __shfl_xor(lsum, o, 32);
  if (lane == 0) sRed[wave] = lsum;
  __syncthreads();
  float gsum = sRed[0];
  for (int w2 = 1; w2 < 8; ++w2) gsum += sRed[w2];

  const float lse = gmax + __logf(gsum);
  for (int n = tid; n < N_; n += 256)
    out[(size_t)b * N_ + n] = sLogit[n] - lse;
}

// ---------------------------------------------------------------------------
// Host launch
// ---------------------------------------------------------------------------
extern "C" void kernel_launch(void* const* d_in, const int* in_sizes, int n_in,
                              void* d_out, int out_size, void* d_ws, size_t ws_size,
                              hipStream_t stream) {
  (void)in_sizes; (void)n_in; (void)out_size; (void)ws_size;
  const float*         emb  = (const float*)d_in[0];
  const float*         ctx  = (const float*)d_in[1];
  const float*         stp  = (const float*)d_in[2];
  const unsigned char* mask = (const unsigned char*)d_in[3]; // jnp bool
  const float* Wqc = (const float*)d_in[4];
  const float* bqc = (const float*)d_in[5];
  const float* Wqs = (const float*)d_in[6];
  const float* bqs = (const float*)d_in[7];
  const float* Wk  = (const float*)d_in[8];
  const float* bk  = (const float*)d_in[9];
  const float* Wkt = (const float*)d_in[10];
  const float* bkt = (const float*)d_in[11];
  const float* Wv  = (const float*)d_in[12];
  const float* bv  = (const float*)d_in[13];
  const float* Wo  = (const float*)d_in[14];
  const float* bo  = (const float*)d_in[15];
  float* out = (float*)d_out;

  // workspace layout (floats): Q | K | V | K_tanh | E_f16 | W_f16
  float* ws  = (float*)d_ws;
  float* Qw  = ws;
  float* Kw  = Qw + (size_t)B_ * D_;
  float* Vw  = Kw + (size_t)B_ * N_ * D_;
  float* KTw = Vw + (size_t)B_ * N_ * D_;
  _Float16* Ef16 = (_Float16*)(KTw + (size_t)B_ * N_ * D_);
  _Float16* Wf16 = Ef16 + (size_t)B_ * N_ * D_;

  cvt_kernel<<<1024, 256, 0, stream>>>(emb, Ef16, Wk, Wv, Wkt, Wf16);
  q_kernel<<<B_, 128, 0, stream>>>(ctx, stp, Wqc, bqc, Wqs, bqs, Qw);
  proj_kernel<<<(B_ * N_) / MT, 256, 0, stream>>>(Ef16, Wf16, bk, bv, bkt,
                                                  Kw, Vw, KTw);
  attn_kernel<<<B_, 256, 0, stream>>>(Kw, Vw, KTw, Qw, mask, Wo, bo, out);
}
